// RegionProposalNetwork_41515153883375
// MI455X (gfx1250) — matrix operand order repaired and checked by reading, not compiled, not run on hardware
//
#include <hip/hip_runtime.h>
#include <hip/hip_bf16.h>
#include <math.h>

typedef __bf16 bf16;
typedef __attribute__((ext_vector_type(16))) __bf16 v16bf;
typedef __attribute__((ext_vector_type(8)))  __bf16 v8bf;
typedef __attribute__((ext_vector_type(8)))  float  v8f;

#define BATCH   8
#define CIN     512
#define COUT    512
#define HH      64
#define WW      64
#define PAD     66              // 64 + 1 halo each side
#define HWSZ    4096            // 64*64
#define M_TOTAL 32768           // B*H*W
#define K_TOTAL 4608            // CIN*9
#define KT_CNT  144             // K / 32
#define NT_CNT  32              // COUT / 16
#define NANCH   36864           // HW*9
#define NPOST   300

// ---------------------------------------------------------------------------
// NCHW f32 -> zero-padded NHWC bf16:  xpad[b][yy][xx][ci], yy/xx in 0..65.
// Halo materialized -> no bounds checks in the conv; K (= kidx*512 + ci)
// becomes the contiguous memory axis so A-fragments are b128 loads.
// ---------------------------------------------------------------------------
__global__ void pad_nhwc_bf16(const float* __restrict__ x,
                              bf16* __restrict__ xpad) {
    int i = blockIdx.x * blockDim.x + threadIdx.x;   // 8*66*66*512 exact
    int ci   = i & 511;
    int rest = i >> 9;
    int xx   = rest % PAD;  rest /= PAD;
    int yy   = rest % PAD;
    int b    = rest / PAD;
    float v = 0.f;
    if (yy >= 1 && yy <= HH && xx >= 1 && xx <= WW)
        v = x[(((size_t)b * CIN + ci) * HH + (yy - 1)) * WW + (xx - 1)];
    xpad[i] = (bf16)v;
}

// ---------------------------------------------------------------------------
// Pack conv_w (OIHW f32) into the per-lane WMMA B-fragment layout, bf16.
// K ordering: k = kidx*512 + ci  (kidx = ky*3+kx).
// wpack[((nt*144 + kt)*32 + lane)*16 + e]: element e of lane L is
// B[k = kt*32 + (L<16?0:16) + e][n = nt*16 + (L&15)]  (VGPR j holds K=2j,2j+1),
// i.e. each lane's 16 bf16 are contiguous -> two b128 loads in the conv.
// ---------------------------------------------------------------------------
__global__ void pack_w(const float* __restrict__ wf, bf16* __restrict__ wpack) {
    int i = blockIdx.x * blockDim.x + threadIdx.x;   // 2,359,296 threads exact
    int e    = i & 15;
    int lane = (i >> 4) & 31;
    int rest = i >> 9;
    int kt   = rest % KT_CNT;
    int nt   = rest / KT_CNT;
    int n      = (nt << 4) + (lane & 15);
    int klocal = ((lane < 16) ? 0 : 16) + e;
    int kidx   = kt >> 4;                   // tap index, constant per K-tile
    int ci     = ((kt & 15) << 5) + klocal;
    wpack[i] = (bf16)wf[(n * CIN + ci) * 9 + kidx];
}

// ---------------------------------------------------------------------------
// 3x3 conv as implicit GEMM on v_wmma_f32_16x16x32_bf16 (wave32).
// One wave computes a 16(M) x 64(N) tile: 4 f32 accumulators, each A fragment
// feeds 4 WMMAs (2 A-loads + 8 B-loads + 4 wmma per K-step).  The 8 waves of
// a block share one M-tile -> identical A addresses -> L0 broadcast.
// global_prefetch of the next tap's activation row overlaps the K-loop.
// Epilogue: +bias, ReLU, store bf16 NHWC [m][cout] for the 1x1 heads.
// ---------------------------------------------------------------------------
__global__ __launch_bounds__(256) void conv3x3_wmma(
        const bf16* __restrict__ xpad, const bf16* __restrict__ wpack,
        const float* __restrict__ convb, bf16* __restrict__ hbf) {
    const int lane  = threadIdx.x & 31;
    const int wv    = blockIdx.x * 8 + (threadIdx.x >> 5);
    const int ng    = wv & 7;                  // N-group: nt = ng*4 .. ng*4+3
    const int mt    = wv >> 3;
    const int Ml    = lane & 15;
    const int m0    = mt << 4;
    const int mrow  = m0 + Ml;                 // A-row owned by this lane
    const int b     = mrow >> 12;
    const int y     = (mrow >> 6) & 63;        // padded row = y + ky (ky 0..2)
    const int x     = mrow & 63;               // padded col = x + kx
    const int khalf = (lane < 16) ? 0 : 16;    // A-fragment K split (contiguous k = 16h + s, same as wpack)

    v8f acc[4];
    #pragma unroll
    for (int t = 0; t < 4; ++t)
        acc[t] = (v8f){0.f, 0.f, 0.f, 0.f, 0.f, 0.f, 0.f, 0.f};

    // per-lane base of this N-group's packed weights; tile t at +t*144*512
    const bf16* __restrict__ wbase =
        wpack + (size_t)(ng * 4) * KT_CNT * 512 + lane * 16;

    #pragma unroll
    for (int kidx = 0; kidx < 9; ++kidx) {
        const int ky = kidx / 3;
        const int kx = kidx - 3 * ky;
        const bf16* __restrict__ arow =
            xpad + ((((size_t)b * PAD + (y + ky)) * PAD + (x + kx)) << 9) + khalf;

        if (kidx < 8) {                        // prefetch next tap's A row
            const int k2  = kidx + 1;
            const int ky2 = k2 / 3;
            const int kx2 = k2 - 3 * ky2;
            const bf16* nrow =
                xpad + ((((size_t)b * PAD + (y + ky2)) * PAD + (x + kx2)) << 9);
            __builtin_prefetch(nrow, 0, 0);        // global_prefetch_b8
            __builtin_prefetch(nrow + 256, 0, 0);
        }

        #pragma unroll 2
        for (int c = 0; c < 16; ++c) {         // ci sub-tiles of 32
            const bf16* ap = arow + (c << 5);
            v8bf alo = *(const v8bf*)(ap);        // klocal khalf..khalf+7
            v8bf ahi = *(const v8bf*)(ap + 8);    // klocal khalf+8..khalf+15
            v16bf afrag = __builtin_shufflevector(
                alo, ahi, 0,1,2,3,4,5,6,7,8,9,10,11,12,13,14,15);

            #pragma unroll
            for (int t = 0; t < 4; ++t) {      // reuse A across 4 N-tiles
                const bf16* wp =
                    wbase + ((size_t)t * KT_CNT + (kidx * 16 + c)) * 512;
                v8bf blo = *(const v8bf*)(wp);
                v8bf bhi = *(const v8bf*)(wp + 8);
                v16bf bfrag = __builtin_shufflevector(
                    blo, bhi, 0,1,2,3,4,5,6,7,8,9,10,11,12,13,14,15);
                acc[t] = __builtin_amdgcn_wmma_f32_16x16x32_bf16(
                             false, afrag, false, bfrag, (short)0, acc[t],
                             false, false);
            }
        }
    }

    #pragma unroll
    for (int t = 0; t < 4; ++t) {
        const int n = ((ng * 4 + t) << 4) + Ml;   // C/D: N = lane%16
        const float bias = convb[n];
        #pragma unroll
        for (int r = 0; r < 8; ++r) {             // C/D: M = r + (lane<16?0:8)
            int m = m0 + r + ((lane < 16) ? 0 : 8);
            float v = acc[t][r] + bias;
            v = v > 0.f ? v : 0.f;
            hbf[(size_t)m * COUT + n] = (bf16)v;
        }
    }
}

// ---------------------------------------------------------------------------
// 1x1 heads: loc (36 ch) and cls (18 ch, sigmoid). Writes directly into the
// reference's transposed output layouts; stashes objectness for NMS.
// ---------------------------------------------------------------------------
__global__ void rpn_heads(const bf16* __restrict__ hbf,
                          const float* __restrict__ locw, const float* __restrict__ locb,
                          const float* __restrict__ clsw, const float* __restrict__ clsb,
                          float* __restrict__ out_cls, float* __restrict__ out_loc,
                          float* __restrict__ scores) {
    int tid = blockIdx.x * blockDim.x + threadIdx.x;
    int m  = tid >> 6;
    int ch = tid & 63;
    if (m >= M_TOTAL || ch >= 54) return;
    int b  = m >> 12;
    int hw = m & (HWSZ - 1);
    const bf16* __restrict__ hrow = hbf + (size_t)m * COUT;

    if (ch < 36) {                              // loc head
        const float* __restrict__ wrow = locw + ch * CIN;
        float acc = locb[ch];
        #pragma unroll 4
        for (int k = 0; k < CIN; ++k) acc += (float)hrow[k] * wrow[k];
        int a = ch >> 2, c = ch & 3;            // (36) -> (9,4)
        out_loc[((size_t)b * NANCH + hw * 9 + a) * 4 + c] = acc;
    } else {                                    // cls head
        int cc = ch - 36;
        const float* __restrict__ wrow = clsw + cc * CIN;
        float acc = clsb[cc];
        #pragma unroll 4
        for (int k = 0; k < CIN; ++k) acc += (float)hrow[k] * wrow[k];
        float s = 1.0f / (1.0f + __expf(-acc));
        int a = cc >> 1, c2 = cc & 1;           // (18) -> (9,2)
        out_cls[((size_t)b * NANCH + hw * 9 + a) * 2 + c2] = s;
        if (c2 == 1) scores[(size_t)b * NANCH + hw * 9 + a] = s;  // objectness
    }
}

// ---------------------------------------------------------------------------
// Anchor gen + loc2bbox + clip + min-size mask.
// ---------------------------------------------------------------------------
__global__ void make_proposals(const float* __restrict__ out_loc,
                               float* __restrict__ scores,
                               float* __restrict__ boxes,
                               const int* __restrict__ img_h,
                               const int* __restrict__ img_w) {
    int tid = blockIdx.x * blockDim.x + threadIdx.x;
    if (tid >= BATCH * NANCH) return;
    int b  = tid / NANCH;
    int j  = tid - b * NANCH;
    int hw = j / 9, a = j - hw * 9;
    int y  = hw >> 6, x = hw & 63;

    const float ratios[3]  = {0.5f, 1.0f, 2.0f};
    const float scalesv[3] = {8.0f, 16.0f, 32.0f};
    float r = ratios[a / 3], s = scalesv[a % 3];
    float ah = 16.0f * s * sqrtf(r);
    float aw = 16.0f * s * sqrtf(1.0f / r);
    float cy = y * 16.0f, cx = x * 16.0f;       // base anchors are centered

    const float* __restrict__ l = out_loc + (size_t)tid * 4;
    cy += l[0] * ah;
    cx += l[1] * aw;
    float hh = ah * __expf(l[2]);
    float ww = aw * __expf(l[3]);

    float fh = (float)img_h[0], fw = (float)img_w[0];
    float y1 = fminf(fmaxf(cy - 0.5f * hh, 0.f), fh);
    float x1 = fminf(fmaxf(cx - 0.5f * ww, 0.f), fw);
    float y2 = fminf(fmaxf(cy + 0.5f * hh, 0.f), fh);
    float x2 = fminf(fmaxf(cx + 0.5f * ww, 0.f), fw);

    float* __restrict__ bx = boxes + (size_t)tid * 4;
    bx[0] = y1; bx[1] = x1; bx[2] = y2; bx[3] = x2;
    if (!(((y2 - y1) >= 16.0f) && ((x2 - x1) >= 16.0f)))
        scores[tid] = -INFINITY;
}

// ---------------------------------------------------------------------------
// Greedy NMS: one block per image; 300 rounds of LDS argmax + suppression.
// Tie-break toward lower index to match jnp.argmax.
// ---------------------------------------------------------------------------
__global__ __launch_bounds__(512) void nms_kernel(
        const float* __restrict__ boxes_all, float* __restrict__ scores_all,
        float* __restrict__ rois, int* __restrict__ roi_idx) {
    const int b   = blockIdx.x;
    const int tid = threadIdx.x;
    const float* __restrict__ boxes = boxes_all + (size_t)b * NANCH * 4;
    float* __restrict__ sc = scores_all + (size_t)b * NANCH;

    __shared__ float sv[512];
    __shared__ int   si[512];
    __shared__ float pick[5];

    for (int it = 0; it < NPOST; ++it) {
        float bv = -INFINITY; int bi = 0;
        for (int j = tid; j < NANCH; j += 512) {
            float v = sc[j];
            if (v > bv) { bv = v; bi = j; }
        }
        sv[tid] = bv; si[tid] = bi;
        __syncthreads();
        for (int s = 256; s > 0; s >>= 1) {
            if (tid < s) {
                if (sv[tid + s] > sv[tid] ||
                    (sv[tid + s] == sv[tid] && si[tid + s] < si[tid])) {
                    sv[tid] = sv[tid + s]; si[tid] = si[tid + s];
                }
            }
            __syncthreads();
        }
        if (tid == 0) {
            int i = si[0];
            int valid = (sv[0] > -INFINITY) ? 1 : 0;
            float b0 = boxes[i * 4 + 0], b1 = boxes[i * 4 + 1];
            float b2 = boxes[i * 4 + 2], b3 = boxes[i * 4 + 3];
            pick[0] = b0; pick[1] = b1; pick[2] = b2; pick[3] = b3;
            pick[4] = (float)valid;
            size_t o = ((size_t)b * NPOST + it) * 4;
            rois[o + 0] = valid ? b0 : 0.f;
            rois[o + 1] = valid ? b1 : 0.f;
            rois[o + 2] = valid ? b2 : 0.f;
            rois[o + 3] = valid ? b3 : 0.f;
            roi_idx[b * NPOST + it] = b;
            sc[i] = -INFINITY;
        }
        __syncthreads();
        float p0 = pick[0], p1 = pick[1], p2 = pick[2], p3 = pick[3];
        bool pvalid = pick[4] > 0.5f;
        float areaP = (p2 - p0) * (p3 - p1);
        if (pvalid) {
            for (int j = tid; j < NANCH; j += 512) {
                float q0 = boxes[j * 4 + 0], q1 = boxes[j * 4 + 1];
                float q2 = boxes[j * 4 + 2], q3 = boxes[j * 4 + 3];
                float yy1 = fmaxf(q0, p0), xx1 = fmaxf(q1, p1);
                float yy2 = fminf(q2, p2), xx2 = fminf(q3, p3);
                float inter = fmaxf(yy2 - yy1, 0.f) * fmaxf(xx2 - xx1, 0.f);
                float areaQ = (q2 - q0) * (q3 - q1);
                float iou = inter / (areaQ + areaP - inter + 1e-9f);
                if (iou > 0.7f) sc[j] = -INFINITY;
            }
        }
        __syncthreads();
    }
}

// ---------------------------------------------------------------------------
extern "C" void kernel_launch(void* const* d_in, const int* in_sizes, int n_in,
                              void* d_out, int out_size, void* d_ws, size_t ws_size,
                              hipStream_t stream) {
    const float* x      = (const float*)d_in[0];
    const float* conv_w = (const float*)d_in[1];
    const float* conv_b = (const float*)d_in[2];
    const float* loc_w  = (const float*)d_in[3];
    const float* loc_b  = (const float*)d_in[4];
    const float* cls_w  = (const float*)d_in[5];
    const float* cls_b  = (const float*)d_in[6];
    const int*   img_h  = (const int*)d_in[7];
    const int*   img_w  = (const int*)d_in[8];
    (void)in_sizes; (void)n_in; (void)out_size; (void)ws_size;

    char* ws = (char*)d_ws;
    bf16*  xpad   = (bf16*)(ws);                     // 8*66*66*512*2 = 35,684,352 B
    bf16*  wpack  = (bf16*)(ws + 35684352u);         //  4,718,592 B
    bf16*  hbf    = (bf16*)(ws + 40402944u);         // 33,554,432 B
    float* boxes  = (float*)(ws + 73957376u);        //  4,718,592 B
    float* scores = (float*)(ws + 78675968u);        //  1,179,648 B

    float* out      = (float*)d_out;
    float* out_cls  = out;                   // 8*36864*2 = 589824
    float* out_loc  = out + 589824;          // 8*36864*4 = 1179648
    float* out_rois = out + 1769472;         // 8*300*4   = 9600
    int*   out_idx  = (int*)(out + 1779072); // 8*300 int32

    pad_nhwc_bf16<<<69696, 256, 0, stream>>>(x, xpad);
    pack_w       <<<9216,  256, 0, stream>>>(conv_w, wpack);
    conv3x3_wmma <<<2048,  256, 0, stream>>>(xpad, wpack, conv_b, hbf);
    rpn_heads    <<<8192,  256, 0, stream>>>(hbf, loc_w, loc_b, cls_w, cls_b,
                                             out_cls, out_loc, scores);
    make_proposals<<<(BATCH * NANCH + 255) / 256, 256, 0, stream>>>(
                                             out_loc, scores, boxes, img_h, img_w);
    nms_kernel   <<<BATCH, 512, 0, stream>>>(boxes, scores, out_rois, out_idx);
}
